// MHA_FusedQKV_16836271800665
// MI455X (gfx1250) — matrix operand order repaired and checked by reading, not compiled, not run on hardware
//
#include <hip/hip_runtime.h>
#include <math.h>

typedef __bf16 bf16;
typedef __attribute__((ext_vector_type(16))) __bf16 v16bf;
typedef __attribute__((ext_vector_type(8)))  float  v8f;

#define D_MODEL 1024
#define N_HEADS 16
#define D_HEAD  64
#define SEQ     2048
#define BATCH   4

// softmax logit scale folded into q at the QKV epilogue:
// p = exp((q.k)/sqrt(Dh) - m) == exp2(((q*QSCALE).k) - m2) with
#define QSCALE (0.125f * 1.44269504f)   // 1/sqrt(Dh) * log2(e)

// ---- optional CDNA5 async global->LDS path (guarded; sync fallback) -------
// Probe-confirmed parameter type (from hipcc diagnostic):
//   param0: int __attribute__((vector_size(16))) addrspace(1)* (global src)
//   param1: LDS destination pointer (addrspace(3))
#if __has_builtin(__builtin_amdgcn_global_load_async_to_lds_b128) && \
    __has_builtin(__builtin_amdgcn_s_wait_asynccnt)
#define ASYNC_LDS 1
typedef int v4i_ls __attribute__((vector_size(16)));
typedef __attribute__((address_space(1))) v4i_ls* glb_v4i_p;
typedef __attribute__((address_space(3))) v4i_ls* lds_v4i_p;
#else
#define ASYNC_LDS 0
#endif

// ---------------------------------------------------------------------------
// Fragment helpers (CDNA5 wave32 WMMA 16x16x32 bf16 layouts, ISA 7.12.2)
// ---------------------------------------------------------------------------

// A-fragment (16x32): lane L holds row M=L%16; element e -> K = e + 16*(L>=16).
static __device__ inline v16bf load_a_frag(const bf16* __restrict__ src, int ld) {
  int lane = threadIdx.x & 31;
  const bf16* p = src + (lane & 15) * ld + ((lane >> 4) & 1) * 16;
  v16bf a;
#pragma unroll
  for (int e = 0; e < 8; ++e) {
    a[e]     = p[e];
    a[e + 8] = p[e + 8];
  }
  return a;
}

// 16 contiguous bf16 (32B, 16B-aligned) -> v16bf. B-frag from transposed
// storage Bt[N][K]: lane L reads row N=L%16, K = 16*(L>=16) .. +15.
static __device__ inline v16bf load16(const bf16* __restrict__ p) {
  union { uint4 u[2]; v16bf v; } c;
  c.u[0] = *reinterpret_cast<const uint4*>(p);
  c.u[1] = *reinterpret_cast<const uint4*>(p + 8);
  return c.v;
}

union B16x16 { uint4 u[2]; bf16 h[16]; };

static __device__ inline v8f wmma_bf16(v16bf a, v16bf b, v8f c) {
  return __builtin_amdgcn_wmma_f32_16x16x32_bf16(false, a, false, b, (short)0, c,
                                                 false, false);
}

// ---------------------------------------------------------------------------
// GEMM: C[M][N] = A[M][K] * B[K][N] + bias[N].
// Block tile 128x128, K-step 32. 8 waves as 4(M)x2(N): each wave 32x64 out
// (2 A-frags x 4 B-frags -> 8 WMMA per K-step). Register double-buffered
// staging; fp32->bf16 conversion folded into the LDS store.
//   EPI==0: scatter bf16 into q/k/v laid out [B,H,S,Dh] (q pre-scaled by QSCALE)
//   EPI==1: fp32 + bias row-major.
// ---------------------------------------------------------------------------
template <typename TA, int EPI>
__global__ void __launch_bounds__(256)
gemm_wmma(const TA* __restrict__ A, const float* __restrict__ B,
          const float* __restrict__ bias, int M, int N, int K,
          bf16* __restrict__ q_out, bf16* __restrict__ k_out,
          bf16* __restrict__ v_out, float* __restrict__ c_out) {
  const int ASTR = 40, BSTR = 40;
  __shared__ __align__(16) bf16 As[128 * ASTR];   // [m][k]
  __shared__ __align__(16) bf16 Bs[128 * BSTR];   // transposed: [n][k]

  const int tid  = threadIdx.x;
  const int wave = tid >> 5;
  const int lane = tid & 31;
  const int hb   = (lane >> 4) & 1;
  const int wm   = wave & 3;        // M group: rows wm*32 .. +31
  const int wn   = wave >> 2;       // N group: cols wn*64 .. +63
  const int m0   = blockIdx.y * 128;
  const int n0   = blockIdx.x * 128;

  // staging geometry: A row r=tid/2, 16 elems; B row kr=tid/8, 16 elems
  const int ar = tid >> 1, ac = (tid & 1) * 16;
  const int bk = tid >> 3, bc = (tid & 7) * 16;

  v8f acc[2][4] = {};
  TA    aReg[16];
  float bReg[16];

  {  // prologue: load k0 = 0
    const TA* asrc = A + (size_t)(m0 + ar) * K + ac;
    const float* bsrc = B + (size_t)bk * N + n0 + bc;
#pragma unroll
    for (int j = 0; j < 16; ++j) aReg[j] = asrc[j];
#pragma unroll
    for (int j = 0; j < 16; ++j) bReg[j] = bsrc[j];
  }

  for (int k0 = 0; k0 < K; k0 += 32) {
    {  // commit staged tile to LDS (convert to bf16; B transposed)
      bf16* ad = As + ar * ASTR + ac;
#pragma unroll
      for (int j = 0; j < 16; ++j) ad[j] = (bf16)(float)aReg[j];
#pragma unroll
      for (int j = 0; j < 16; ++j) Bs[(bc + j) * BSTR + bk] = (bf16)bReg[j];
    }
    __syncthreads();

    if (k0 + 32 < K) {  // issue next-tile global loads before compute
      const TA* asrc = A + (size_t)(m0 + ar) * K + (k0 + 32) + ac;
      const float* bsrc = B + (size_t)(k0 + 32 + bk) * N + n0 + bc;
#pragma unroll
      for (int j = 0; j < 16; ++j) aReg[j] = asrc[j];
#pragma unroll
      for (int j = 0; j < 16; ++j) bReg[j] = bsrc[j];
      __builtin_prefetch(asrc + 32, 0, 0);
      __builtin_prefetch(bsrc + (size_t)32 * N, 0, 0);
    }

    v16bf af0 = load_a_frag(As + (wm * 32) * ASTR, ASTR);
    v16bf af1 = load_a_frag(As + (wm * 32 + 16) * ASTR, ASTR);
#pragma unroll
    for (int ni = 0; ni < 4; ++ni) {
      v16bf bf = load16(Bs + (wn * 64 + ni * 16 + (lane & 15)) * BSTR + hb * 16);
      acc[0][ni] = wmma_bf16(af0, bf, acc[0][ni]);
      acc[1][ni] = wmma_bf16(af1, bf, acc[1][ni]);
    }
    __syncthreads();
  }

  // epilogue: C element (row = wm*32 + mi*16 + v + 8*hb, col = wn*64 + ni*16 + lane%16)
#pragma unroll
  for (int mi = 0; mi < 2; ++mi)
#pragma unroll
    for (int ni = 0; ni < 4; ++ni) {
      int col = n0 + wn * 64 + ni * 16 + (lane & 15);
      float bv = bias[col];
#pragma unroll
      for (int v = 0; v < 8; ++v) {
        int gm = m0 + wm * 32 + mi * 16 + v + 8 * hb;
        float val = acc[mi][ni][v] + bv;
        if (EPI == 0) {
          int which = col >> 10;
          int h = (col & 1023) >> 6;
          int d = col & 63;
          int b = gm >> 11;
          int s = gm & 2047;
          size_t dst = (((size_t)b * N_HEADS + h) * SEQ + s) * D_HEAD + d;
          if (which == 0)      q_out[dst] = (bf16)(val * QSCALE);  // fold softmax scale
          else if (which == 1) k_out[dst] = (bf16)val;
          else                 v_out[dst] = (bf16)val;
        } else {
          c_out[(size_t)gm * N + col] = val;
        }
      }
    }
}

// ---------------------------------------------------------------------------
// Flash attention: one block per (128 queries, head, batch); 8 waves x 16 q.
// 64-key chunks. K tile: LDS double-buffered, filled with async global->LDS
// copies (overlapped with compute) when available. V tile: register
// double-buffered, stored transposed. Online softmax in fp32, exp2 domain
// (q pre-scaled by 1/sqrt(Dh)*log2e at the QKV epilogue).
// ---------------------------------------------------------------------------
static const int KSTR = 72;   // Ks row stride (bf16): 64 keys x 64 d (+pad)
static const int VSTR = 72;   // Vt row stride: 64 d x 64 keys (+pad)
static const int PSTR = 72;   // P row stride: 16 q x 64 keys (+pad)
#define KBUF (64 * KSTR)

static __device__ inline void stage_k(const bf16* __restrict__ kh, int kc,
                                      bf16* __restrict__ dstbase, int tid) {
  int key = tid >> 2, d0 = (tid & 3) * 16;
  const bf16* src = kh + (size_t)(kc + key) * D_HEAD + d0;
  bf16* dst = dstbase + key * KSTR + d0;
#if ASYNC_LDS
  __builtin_amdgcn_global_load_async_to_lds_b128((glb_v4i_p)src, (lds_v4i_p)dst,
                                                 0, 0);
  __builtin_amdgcn_global_load_async_to_lds_b128((glb_v4i_p)(src + 8),
                                                 (lds_v4i_p)(dst + 8), 0, 0);
#else
#pragma unroll
  for (int j = 0; j < 16; ++j) dst[j] = src[j];
#endif
}

__global__ void __launch_bounds__(256)
flash_attn(const bf16* __restrict__ Q, const bf16* __restrict__ Kb,
           const bf16* __restrict__ Vb, bf16* __restrict__ Y) {
  __shared__ __align__(16) bf16 Ks[2 * KBUF];      // double-buffered [key][d]
  __shared__ __align__(16) bf16 Vt[64 * VSTR];     // transposed [d][key]
  __shared__ __align__(16) bf16 Ps[8 * 16 * PSTR]; // per-wave P bounce

  const int tid  = threadIdx.x;
  const int wave = tid >> 5;
  const int lane = tid & 31;
  const int hb   = (lane >> 4) & 1;
  const int b    = blockIdx.z, h = blockIdx.y;
  const int q0   = blockIdx.x * 128 + wave * 16;

  const size_t head_off = ((size_t)b * N_HEADS + h) * SEQ * D_HEAD;
  const bf16* qh = Q  + head_off;
  const bf16* kh = Kb + head_off;
  const bf16* vh = Vb + head_off;

  // Q fragments (Dh chunks 0..31, 32..63) held in registers for entire kernel.
  v16bf qf[2];
  {
    const bf16* p = qh + (size_t)(q0 + (lane & 15)) * D_HEAD + hb * 16;
#pragma unroll
    for (int kk = 0; kk < 2; ++kk)
#pragma unroll
      for (int e = 0; e < 8; ++e) {
        qf[kk][e]     = p[kk * 32 + e];
        qf[kk][e + 8] = p[kk * 32 + e + 8];
      }
  }

  v8f accy[4] = {};
  float rmax[8], rsum[8];
#pragma unroll
  for (int v = 0; v < 8; ++v) { rmax[v] = -1e30f; rsum[v] = 0.0f; }

  bf16* Pw = Ps + wave * (16 * PSTR);

  const int vkey = tid >> 2, vd0 = (tid & 3) * 16;  // V staging geometry
  B16x16 vr;

  // prologue: chunk 0 -> K buffer 0 (async if available) + V regs
  stage_k(kh, 0, Ks, tid);
  {
    const bf16* src = vh + (size_t)vkey * D_HEAD + vd0;
    vr.u[0] = *reinterpret_cast<const uint4*>(src);
    vr.u[1] = *reinterpret_cast<const uint4*>(src + 8);
  }

  for (int it = 0; it < SEQ / 64; ++it) {
    const int kc = it * 64;
    const bf16* kbuf = Ks + (it & 1) * KBUF;

#if ASYNC_LDS
    __builtin_amdgcn_s_wait_asynccnt(0);  // my K copies for this buffer done
#endif
    // commit V regs transposed into Vt
#pragma unroll
    for (int j = 0; j < 16; ++j) Vt[(vd0 + j) * VSTR + vkey] = vr.h[j];
    __syncthreads();  // all K copies + V stores visible

    if (kc + 64 < SEQ) {  // prefetch next chunk: async K into idle buffer, V into regs
      stage_k(kh, kc + 64, Ks + ((it + 1) & 1) * KBUF, tid);
      const bf16* src = vh + (size_t)(kc + 64 + vkey) * D_HEAD + vd0;
      vr.u[0] = *reinterpret_cast<const uint4*>(src);
      vr.u[1] = *reinterpret_cast<const uint4*>(src + 8);
    }

    // ---- scores (already in exp2 logit domain): 4 key-tiles x 2 WMMA ----
    v8f sc[4];
#pragma unroll
    for (int nt = 0; nt < 4; ++nt) {
      v16bf bk = load16(kbuf + (nt * 16 + (lane & 15)) * KSTR + hb * 16);
      v8f z = {};
      sc[nt] = wmma_bf16(qf[0], bk, z);
    }
#pragma unroll
    for (int nt = 0; nt < 4; ++nt) {
      v16bf bk = load16(kbuf + (nt * 16 + (lane & 15)) * KSTR + hb * 16 + 32);
      sc[nt] = wmma_bf16(qf[1], bk, sc[nt]);
    }

    // ---- online softmax (rows are half-groups: M = v + 8*hb) ----
    float scale_f[8];
#pragma unroll
    for (int v = 0; v < 8; ++v) {
      float t = fmaxf(fmaxf(sc[0][v], sc[1][v]), fmaxf(sc[2][v], sc[3][v]));
      t = fmaxf(t, __shfl_xor(t, 1, 32));
      t = fmaxf(t, __shfl_xor(t, 2, 32));
      t = fmaxf(t, __shfl_xor(t, 4, 32));
      t = fmaxf(t, __shfl_xor(t, 8, 32));
      float mnew = fmaxf(rmax[v], t);
      scale_f[v] = exp2f(rmax[v] - mnew);
      rmax[v] = mnew;
    }

    float psum[8];
#pragma unroll
    for (int v = 0; v < 8; ++v) psum[v] = 0.0f;
#pragma unroll
    for (int nt = 0; nt < 4; ++nt)
#pragma unroll
      for (int v = 0; v < 8; ++v) {
        float p = exp2f(sc[nt][v] - rmax[v]);
        psum[v] += p;
        Pw[(v + 8 * hb) * PSTR + nt * 16 + (lane & 15)] = (bf16)p;
      }
#pragma unroll
    for (int v = 0; v < 8; ++v) {
      float t = psum[v];
      t += __shfl_xor(t, 1, 32);
      t += __shfl_xor(t, 2, 32);
      t += __shfl_xor(t, 4, 32);
      t += __shfl_xor(t, 8, 32);
      rsum[v] = rsum[v] * scale_f[v] + t;
    }
#pragma unroll
    for (int nd = 0; nd < 4; ++nd)
#pragma unroll
      for (int v = 0; v < 8; ++v) accy[nd][v] *= scale_f[v];

    // ---- y += P @ V : 2 P-frags (32 keys each) x 4 d-tiles ----
#pragma unroll
    for (int pi = 0; pi < 2; ++pi) {
      v16bf pf = load_a_frag(Pw + pi * 32, PSTR);
#pragma unroll
      for (int nd = 0; nd < 4; ++nd) {
        v16bf bv =
            load16(Vt + (nd * 16 + (lane & 15)) * VSTR + hb * 16 + pi * 32);
        accy[nd] = wmma_bf16(pf, bv, accy[nd]);
      }
    }
    __syncthreads();  // Vt reusable next iteration
  }

  // normalize; write y in [B, S, H*Dh] bf16 (out-proj input layout)
#pragma unroll
  for (int nd = 0; nd < 4; ++nd)
#pragma unroll
    for (int v = 0; v < 8; ++v) {
      int srow = q0 + v + 8 * hb;
      int col  = h * D_HEAD + nd * 16 + (lane & 15);
      Y[((size_t)b * SEQ + srow) * D_MODEL + col] = (bf16)(accy[nd][v] / rsum[v]);
    }
}

// ---------------------------------------------------------------------------
extern "C" void kernel_launch(void* const* d_in, const int* in_sizes, int n_in,
                              void* d_out, int out_size, void* d_ws, size_t ws_size,
                              hipStream_t stream) {
  (void)in_sizes; (void)n_in; (void)out_size; (void)ws_size;
  const float* x    = (const float*)d_in[0];
  const float* Wqkv = (const float*)d_in[1];
  const float* bqkv = (const float*)d_in[2];
  const float* Wout = (const float*)d_in[3];
  const float* bout = (const float*)d_in[4];
  float* out = (float*)d_out;

  const size_t elems = (size_t)BATCH * N_HEADS * SEQ * D_HEAD;  // 8M
  bf16* qb = (bf16*)d_ws;
  bf16* kb = qb + elems;
  bf16* vb = kb + elems;
  bf16* yb = vb + elems;  // [B, S, D_MODEL] bf16

  dim3 blk(256);
  gemm_wmma<float, 0><<<dim3(3 * D_MODEL / 128, BATCH * SEQ / 128), blk, 0, stream>>>(
      x, Wqkv, bqkv, BATCH * SEQ, 3 * D_MODEL, D_MODEL, qb, kb, vb, nullptr);
  flash_attn<<<dim3(SEQ / 128, N_HEADS, BATCH), blk, 0, stream>>>(qb, kb, vb, yb);
  gemm_wmma<bf16, 1><<<dim3(D_MODEL / 128, BATCH * SEQ / 128), blk, 0, stream>>>(
      yb, Wout, bout, BATCH * SEQ, D_MODEL, D_MODEL, nullptr, nullptr, nullptr, out);
}
